// UResNet_58085137711793
// MI455X (gfx1250) — compile-verified, matrix operand order checked by reading, not run on hardware
//
#include <hip/hip_runtime.h>

typedef __attribute__((ext_vector_type(16))) __bf16 v16bf;
typedef __attribute__((ext_vector_type(8)))  float  v8f;
typedef __attribute__((ext_vector_type(4)))  unsigned v4u;
typedef __attribute__((ext_vector_type(4)))  int   v4i;
typedef __attribute__((ext_vector_type(8)))  int   v8i;

#define NPTS 100000
#define BN_EPS 1e-4f

#if __has_builtin(__builtin_amdgcn_tensor_load_to_lds) && \
    __has_builtin(__builtin_amdgcn_s_wait_tensorcnt)
#define HAVE_TDM 1
#endif

// K-pair start for 16-bit A/B fragments (CDNA5 16x16x32 layout):
// lanes 0-15 (half=0): VGPR v holds K = {0,2,4,6,16,18,20,22}[v] .. +1
// lanes 16-31 (half=1): same +8
__device__ __forceinline__ int kpair(int v8, int half) {
  return ((v8 < 4) ? (2 * v8) : (16 + 2 * (v8 - 4))) + 8 * half;
}

union BfPair {
  unsigned u;
  __bf16 h[2];
};

// ---------------- scatter points into dense grid ----------------
__global__ void k_scatter(const int* __restrict__ coords, const float* __restrict__ feats,
                          float* __restrict__ dense, float* __restrict__ mask) {
  int p = blockIdx.x * blockDim.x + threadIdx.x;
  if (p >= NPTS) return;
  int x = coords[3 * p], y = coords[3 * p + 1], z = coords[3 * p + 2];
  int v = (x << 12) + (y << 6) + z;
  dense[v] = feats[p];
  mask[v] = 1.0f;
}

// ---------------- input conv 3x3x3, 1 -> 16 ch (padded to 32) ----------------
__global__ void k_conv_in(const float* __restrict__ dense, const float* __restrict__ mask,
                          const float* __restrict__ w, __bf16* __restrict__ out) {
  int v = blockIdx.x * blockDim.x + threadIdx.x;
  if (v >= 64 * 64 * 64) return;
  int vz = v & 63, vy = (v >> 6) & 63, vx = v >> 12;
  float acc[16];
#pragma unroll
  for (int c = 0; c < 16; ++c) acc[c] = 0.f;
  for (int t = 0; t < 27; ++t) {
    int dx = t / 9 - 1, dy = (t / 3) % 3 - 1, dz = t % 3 - 1;
    int ix = vx + dx, iy = vy + dy, iz = vz + dz;
    if ((unsigned)ix < 64u && (unsigned)iy < 64u && (unsigned)iz < 64u) {
      float xin = dense[(ix << 12) + (iy << 6) + iz];
#pragma unroll
      for (int c = 0; c < 16; ++c) acc[c] = fmaf(xin, w[t * 16 + c], acc[c]);
    }
  }
  float mo = mask[v];
  __bf16* o = out + (size_t)v * 32;
#pragma unroll
  for (int c = 0; c < 16; ++c) o[c] = (__bf16)(acc[c] * mo);
#pragma unroll
  for (int c = 16; c < 32; ++c) o[c] = (__bf16)0.f;
}

// ---------------- mask max-pool 2x ----------------
__global__ void k_maskpool(const float* __restrict__ fine, float* __restrict__ coarse, int cs) {
  int v = blockIdx.x * blockDim.x + threadIdx.x;
  int V = cs * cs * cs;
  if (v >= V) return;
  int cz = v % cs, cy = (v / cs) % cs, cx = v / (cs * cs);
  int fs = cs * 2;
  float mx = 0.f;
#pragma unroll
  for (int a = 0; a < 2; ++a)
#pragma unroll
    for (int b = 0; b < 2; ++b)
#pragma unroll
      for (int c = 0; c < 2; ++c)
        mx = fmaxf(mx, fine[((2 * cx + a) * fs + 2 * cy + b) * fs + 2 * cz + c]);
  coarse[v] = mx;
}

// ---------------- mask sum ----------------
__global__ void k_mask_sum(const float* __restrict__ mask, int V, float* __restrict__ cnt) {
  int i = blockIdx.x * blockDim.x + threadIdx.x;
  int stride = gridDim.x * blockDim.x;
  float s = 0.f;
  for (int v = i; v < V; v += stride) s += mask[v];
  atomicAdd(cnt, s);
}

// ---------------- per-channel sum / sumsq (inputs zero outside mask) ----------------
__global__ void k_bn_stats(const __bf16* __restrict__ buf, int V, int Cp,
                           float* __restrict__ sum, float* __restrict__ sumsq) {
  int c = threadIdx.x;
  if (c >= Cp) return;
  int v0 = blockIdx.x * 256;
  int v1 = v0 + 256;
  if (v1 > V) v1 = V;
  float s = 0.f, q = 0.f;
  for (int v = v0; v < v1; ++v) {
    float x = (float)buf[(size_t)v * Cp + c];
    s += x;
    q += x * x;
  }
  atomicAdd(&sum[c], s);
  atomicAdd(&sumsq[c], q);
}

// ---------------- fold BN into per-channel scale/bias ----------------
__global__ void k_bn_finalize(const float* __restrict__ sum, const float* __restrict__ sumsq,
                              const float* __restrict__ cnt, float cconst,
                              const float* __restrict__ g, const float* __restrict__ b,
                              int C, int Cp, float* __restrict__ scale, float* __restrict__ bias) {
  int c = threadIdx.x;
  if (c >= Cp) return;
  if (c < C) {
    float n = cnt ? *cnt : cconst;
    n = fmaxf(n, 1.0f);
    float mu = sum[c] / n;
    float var = sumsq[c] / n - mu * mu;
    float sc = g[c] * rsqrtf(var + BN_EPS);
    scale[c] = sc;
    bias[c] = b[c] - mu * sc;
  } else {
    scale[c] = 0.f;
    bias[c] = 0.f;
  }
}

// ---------------- repack f32 weights [t][cin][cout] -> bf16 [t][coutPad][cinPad] ----------------
__global__ void k_repack(const float* __restrict__ src, __bf16* __restrict__ dst,
                         int taps, int cin, int cout, int cinPad, int coutPad) {
  int idx = blockIdx.x * blockDim.x + threadIdx.x;
  int total = taps * coutPad * cinPad;
  if (idx >= total) return;
  int ci = idx % cinPad;
  int co = (idx / cinPad) % coutPad;
  int t = idx / (cinPad * coutPad);
  float v = (ci < cin && co < cout) ? src[((size_t)t * cin + ci) * cout + co] : 0.f;
  dst[idx] = (__bf16)v;
}

// ---------------- fused BN+relu+mask -> implicit-GEMM conv via WMMA ----------------
// one wave computes 16 voxels x 32 output channels; A fragment shared by 2 N-tiles
template <int K, int PAD, int STRIDE, int KC>
__global__ void __launch_bounds__(32)
k_conv_wmma(const __bf16* __restrict__ in, const __bf16* __restrict__ wt,
            const float* __restrict__ scale, const float* __restrict__ bias,
            const float* __restrict__ maskIn, const float* __restrict__ maskOut,
            const __bf16* __restrict__ residual, __bf16* __restrict__ out,
            int inS, int outS) {
  constexpr int CinP = KC * 32;
  constexpr int TAPS = K * K * K;
  const int CoutP = (int)gridDim.y << 5;
  const int lane = threadIdx.x;
  const int m = lane & 15, half = lane >> 4;
  const int vOut = blockIdx.x * 16 + m;
  const int S2 = outS * outS;
  const int vx = vOut / S2, vy = (vOut / outS) % outS, vz = vOut % outS;
  const int nBase = blockIdx.y * 32;
  v8f acc0 = {}, acc1 = {};
  for (int t = 0; t < TAPS; ++t) {
    int dx = t / (K * K) - PAD, dy = (t / K) % K - PAD, dz = t % K - PAD;
    int ix = vx * STRIDE + dx, iy = vy * STRIDE + dy, iz = vz * STRIDE + dz;
    bool ok = (unsigned)ix < (unsigned)inS && (unsigned)iy < (unsigned)inS &&
              (unsigned)iz < (unsigned)inS;
    int nv = ok ? ((ix * inS + iy) * inS + iz) : 0;
    float mIn = ok ? maskIn[nv] : 0.f;
    const char* aRow = (const char*)(in + (size_t)nv * CinP);
    const char* w0 = (const char*)(wt + ((size_t)t * CoutP + nBase + m) * CinP);
    const char* w1 = w0 + (size_t)16 * CinP * 2;
    __builtin_prefetch(aRow, 0, 1);
#pragma unroll
    for (int kc = 0; kc < KC; ++kc) {
      v16bf a, b0, b1;
#pragma unroll
      for (int v8 = 0; v8 < 8; ++v8) {
        int k0 = kc * 32 + kpair(v8, half);
        BfPair av, p0, p1;
        av.u = ok ? *(const unsigned*)(aRow + 2 * k0) : 0u;
        float y0 = fmaf((float)av.h[0], scale[k0], bias[k0]);
        float y1 = fmaf((float)av.h[1], scale[k0 + 1], bias[k0 + 1]);
        y0 = fmaxf(y0, 0.f);
        y1 = fmaxf(y1, 0.f);
        a[2 * v8] = (__bf16)(y0 * mIn);
        a[2 * v8 + 1] = (__bf16)(y1 * mIn);
        p0.u = *(const unsigned*)(w0 + 2 * k0);
        p1.u = *(const unsigned*)(w1 + 2 * k0);
        b0[2 * v8] = p0.h[0];
        b0[2 * v8 + 1] = p0.h[1];
        b1[2 * v8] = p1.h[0];
        b1[2 * v8 + 1] = p1.h[1];
      }
      acc0 = __builtin_amdgcn_wmma_f32_16x16x32_bf16(false, a, false, b0, (short)0, acc0,
                                                     false, false);
      acc1 = __builtin_amdgcn_wmma_f32_16x16x32_bf16(false, a, false, b1, (short)0, acc1,
                                                     false, false);
    }
  }
#pragma unroll
  for (int i = 0; i < 8; ++i) {
    int row = blockIdx.x * 16 + (half ? 8 : 0) + i;
    float mo = maskOut[row];
    int c0 = nBase + m;
    size_t oi0 = (size_t)row * CoutP + c0;
    size_t oi1 = oi0 + 16;
    float r0 = acc0[i] * mo;
    float r1 = acc1[i] * mo;
    if (residual) {
      r0 += (float)residual[oi0];
      r1 += (float)residual[oi1];
    }
    out[oi0] = (__bf16)r0;
    out[oi1] = (__bf16)r1;
  }
}

// ---------------- fused BN+leaky -> dense GEMM via WMMA (point space) ----------------
// 8 waves per block; 32xKpad bf16 weight tile staged into LDS via the Tensor Data Mover
template <int KC>
__global__ void __launch_bounds__(256)
k_gemm_wmma(const __bf16* __restrict__ A, const __bf16* __restrict__ W,
            const float* __restrict__ scale, const float* __restrict__ bias, float leak,
            __bf16* __restrict__ outB, float* __restrict__ outF,
            const float* __restrict__ outBias, int Npad, int Mtiles) {
  constexpr int Kpad = KC * 32;
  __shared__ __bf16 ldsW[32 * 256];  // up to 32 cols x 256 K of bf16 (16 KB)
  const int nBase = blockIdx.y * 32;
  const int nCols = (Npad - nBase) >= 32 ? 32 : (Npad - nBase);
  const __bf16* wSrc = W + (size_t)nBase * Kpad;
#ifdef HAVE_TDM
  if ((threadIdx.x >> 5) == 0) {  // wave 0 issues one TDM descriptor
    unsigned long long ga = (unsigned long long)(size_t)wSrc;
    unsigned ldsOff = (unsigned)(unsigned long long)(size_t)&ldsW[0];
    v4u g0;
    g0[0] = 1u;                                             // count=1 (valid user D#)
    g0[1] = ldsOff;                                         // lds_addr
    g0[2] = (unsigned)ga;                                   // global_addr[31:0]
    g0[3] = ((unsigned)(ga >> 32) & 0x01FFFFFFu) | (2u << 30);  // addr[56:32] | type=2
    v8i g1;
    g1[0] = 0x00010000;                    // data_size = 2 bytes
    g1[1] = (int)((unsigned)Kpad << 16);   // tensor_dim0[15:0]
    g1[2] = (int)((unsigned)nCols << 16);  // tensor_dim1[15:0]
    g1[3] = (int)((unsigned)Kpad << 16);   // tile_dim0
    g1[4] = (int)(unsigned)nCols;          // tile_dim1 (tile_dim2 = 0)
    g1[5] = (int)(unsigned)Kpad;           // tensor_dim0_stride[31:0]
    g1[6] = 0;
    g1[7] = 0;
    v4i z4 = {0, 0, 0, 0};
#if __clang_major__ >= 23
    v8i z8 = {0, 0, 0, 0, 0, 0, 0, 0};
    __builtin_amdgcn_tensor_load_to_lds(g0, g1, z4, z4, z8, 0);
#else
    __builtin_amdgcn_tensor_load_to_lds(g0, g1, z4, z4, 0);
#endif
    __builtin_amdgcn_s_wait_tensorcnt(0);
  }
#else
  for (int c = threadIdx.x; c < (nCols * Kpad) / 8; c += 256)
    ((uint4*)ldsW)[c] = ((const uint4*)wSrc)[c];
#endif
  __syncthreads();

  const int wave = threadIdx.x >> 5;
  const int lane = threadIdx.x & 31;
  const int m = lane & 15, half = lane >> 4;
  const int tile = blockIdx.x * 8 + wave;
  if (tile >= Mtiles) return;
  const bool hasN2 = (nBase + 16) < Npad;
  const char* aRow = (const char*)(A + ((size_t)tile * 16 + m) * Kpad);
  const char* l0 = (const char*)&ldsW[(size_t)m * Kpad];
  const char* l1 = (const char*)&ldsW[(size_t)(m + 16) * Kpad];
  v8f acc0 = {}, acc1 = {};
#pragma unroll
  for (int kc = 0; kc < KC; ++kc) {
    __builtin_prefetch(aRow + kc * 64 + 64, 0, 1);
    v16bf a, b0, b1;
#pragma unroll
    for (int v8 = 0; v8 < 8; ++v8) {
      int k0 = kc * 32 + kpair(v8, half);
      BfPair av, p0, p1;
      av.u = *(const unsigned*)(aRow + 2 * k0);
      float y0 = fmaf((float)av.h[0], scale[k0], bias[k0]);
      float y1 = fmaf((float)av.h[1], scale[k0 + 1], bias[k0 + 1]);
      y0 = (y0 > 0.f) ? y0 : leak * y0;
      y1 = (y1 > 0.f) ? y1 : leak * y1;
      a[2 * v8] = (__bf16)y0;
      a[2 * v8 + 1] = (__bf16)y1;
      p0.u = *(const unsigned*)(l0 + 2 * k0);
      b0[2 * v8] = p0.h[0];
      b0[2 * v8 + 1] = p0.h[1];
      if (hasN2) {
        p1.u = *(const unsigned*)(l1 + 2 * k0);
        b1[2 * v8] = p1.h[0];
        b1[2 * v8 + 1] = p1.h[1];
      }
    }
    acc0 = __builtin_amdgcn_wmma_f32_16x16x32_bf16(false, a, false, b0, (short)0, acc0,
                                                   false, false);
    if (hasN2)
      acc1 = __builtin_amdgcn_wmma_f32_16x16x32_bf16(false, a, false, b1, (short)0, acc1,
                                                     false, false);
  }
#pragma unroll
  for (int i = 0; i < 8; ++i) {
    size_t r = (size_t)tile * 16 + (half ? 8 : 0) + i;
    int c0 = nBase + m;
    if (outF) {
      outF[r * 16 + c0] = acc0[i] + outBias[c0];
    } else {
      outB[r * (size_t)Npad + c0] = (__bf16)acc0[i];
      if (hasN2) outB[r * (size_t)Npad + c0 + 16] = (__bf16)acc1[i];
    }
  }
}

// ---------------- multi-scale gather + concat into point buffer [p][256] ----------------
__global__ void k_gather_concat(const int* __restrict__ coords, const __bf16* __restrict__ f0,
                                const __bf16* __restrict__ f1, const __bf16* __restrict__ f2,
                                const __bf16* __restrict__ f3, const __bf16* __restrict__ f4,
                                __bf16* __restrict__ pts) {
  int p = blockIdx.x * blockDim.x + threadIdx.x;
  if (p >= NPTS) return;
  int x = coords[3 * p], y = coords[3 * p + 1], z = coords[3 * p + 2];
  __bf16* dst = pts + (size_t)p * 256;
  int o = 0;
  {
    const __bf16* s = f4 + (size_t)(((x >> 4) * 4 + (y >> 4)) * 4 + (z >> 4)) * 96;
    for (int c = 0; c < 80; ++c) dst[o++] = s[c];
  }
  {
    const __bf16* s = f3 + (size_t)(((x >> 3) * 8 + (y >> 3)) * 8 + (z >> 3)) * 64;
    for (int c = 0; c < 64; ++c) dst[o++] = s[c];
  }
  {
    const __bf16* s = f2 + (size_t)(((x >> 2) * 16 + (y >> 2)) * 16 + (z >> 2)) * 64;
    for (int c = 0; c < 48; ++c) dst[o++] = s[c];
  }
  {
    const __bf16* s = f1 + (size_t)(((x >> 1) * 32 + (y >> 1)) * 32 + (z >> 1)) * 32;
    for (int c = 0; c < 32; ++c) dst[o++] = s[c];
  }
  {
    const __bf16* s = f0 + (size_t)((x * 64 + y) * 64 + z) * 32;
    for (int c = 0; c < 16; ++c) dst[o++] = s[c];
  }
  for (; o < 256; ++o) dst[o] = (__bf16)0.f;
}

// =================================================================
extern "C" void kernel_launch(void* const* d_in, const int* in_sizes, int n_in,
                              void* d_out, int out_size, void* d_ws, size_t ws_size,
                              hipStream_t stream) {
  (void)in_sizes; (void)n_in; (void)out_size; (void)ws_size;
  const int Sl[5] = {64, 32, 16, 8, 4};
  const int Vl[5] = {262144, 32768, 4096, 512, 64};
  const int Cl[5] = {16, 32, 48, 64, 80};
  const int Cpl[5] = {32, 32, 64, 64, 96};

  const int* coords = (const int*)d_in[0];
  const float* feats = (const float*)d_in[1];
  const float* w_in = (const float*)d_in[2];
  auto enc = [&](int i, int r, int j) { return (const float*)d_in[3 + (i * 2 + r) * 6 + j]; };
  auto dwn = [&](int i, int j) { return (const float*)d_in[63 + i * 3 + j]; };
  auto dcw = [&](int i, int j) { return (const float*)d_in[75 + i * 3 + j]; };
  const float* out_g = (const float*)d_in[87];
  const float* out_b = (const float*)d_in[88];
  const float* lin_w = (const float*)d_in[89];
  const float* lin_b = (const float*)d_in[90];

  char* base = (char*)d_ws;
  size_t off = 0;
  auto alloc = [&](size_t bytes) -> char* {
    char* p = base + off;
    off += (bytes + 255) & ~(size_t)255;
    return p;
  };
  float* dense0 = (float*)alloc((size_t)Vl[0] * 4);
  float* mask[5];
  for (int i = 0; i < 5; ++i) mask[i] = (float*)alloc((size_t)Vl[i] * 4);
  __bf16* fmap[5];
  for (int i = 0; i < 5; ++i) fmap[i] = (__bf16*)alloc((size_t)Vl[i] * Cpl[i] * 2);
  __bf16* tmp = (__bf16*)alloc((size_t)Vl[0] * Cpl[0] * 2);
  __bf16* wrep = (__bf16*)alloc((size_t)27 * 96 * 96 * 2);
  float* bnSum = (float*)alloc(2080);
  float* bnSq = bnSum + 256;
  float* bnCnt = bnSum + 512;
  float* bnScale = (float*)alloc(2048);
  float* bnBias = bnScale + 256;
  __bf16* ptsA = (__bf16*)alloc((size_t)NPTS * 256 * 2);
  __bf16* ptsB = (__bf16*)alloc((size_t)NPTS * 256 * 2);

  hipMemsetAsync(dense0, 0, (size_t)Vl[0] * 4, stream);
  hipMemsetAsync(mask[0], 0, (size_t)Vl[0] * 4, stream);
  k_scatter<<<(NPTS + 255) / 256, 256, 0, stream>>>(coords, feats, dense0, mask[0]);
  k_conv_in<<<(Vl[0] + 255) / 256, 256, 0, stream>>>(dense0, mask[0], w_in, fmap[0]);

  auto runBN = [&](const __bf16* buf, int V, int Cp, int C, const float* g, const float* b,
                   const float* mk) {
    hipMemsetAsync(bnSum, 0, 2080, stream);
    if (mk) k_mask_sum<<<64, 256, 0, stream>>>(mk, V, bnCnt);
    k_bn_stats<<<(V + 255) / 256, 256, 0, stream>>>(buf, V, Cp, bnSum, bnSq);
    k_bn_finalize<<<1, 256, 0, stream>>>(bnSum, bnSq, mk ? bnCnt : (const float*)nullptr,
                                         (float)NPTS, g, b, C, Cp, bnScale, bnBias);
  };
  auto runConv = [&](const __bf16* in, const float* wsrc, int K, int Cin, int Cout, int CinP,
                     int CoutP, int inS, int outS, int Vout, const float* mIn, const float* mOut,
                     const __bf16* res, __bf16* out) {
    int taps = K * K * K;
    int total = taps * CoutP * CinP;
    k_repack<<<(total + 255) / 256, 256, 0, stream>>>(wsrc, wrep, taps, Cin, Cout, CinP, CoutP);
    dim3 g(Vout / 16, CoutP / 32);
    int KC = CinP / 32;
    if (K == 3) {
      if (KC == 1)
        k_conv_wmma<3, 1, 1, 1><<<g, 32, 0, stream>>>(in, wrep, bnScale, bnBias, mIn, mOut, res,
                                                      out, inS, outS);
      else if (KC == 2)
        k_conv_wmma<3, 1, 1, 2><<<g, 32, 0, stream>>>(in, wrep, bnScale, bnBias, mIn, mOut, res,
                                                      out, inS, outS);
      else
        k_conv_wmma<3, 1, 1, 3><<<g, 32, 0, stream>>>(in, wrep, bnScale, bnBias, mIn, mOut, res,
                                                      out, inS, outS);
    } else {
      if (KC == 1)
        k_conv_wmma<2, 0, 2, 1><<<g, 32, 0, stream>>>(in, wrep, bnScale, bnBias, mIn, mOut, res,
                                                      out, inS, outS);
      else
        k_conv_wmma<2, 0, 2, 2><<<g, 32, 0, stream>>>(in, wrep, bnScale, bnBias, mIn, mOut, res,
                                                      out, inS, outS);
    }
  };

  for (int i = 0; i < 5; ++i) {
    for (int r = 0; r < 2; ++r) {
      runBN(fmap[i], Vl[i], Cpl[i], Cl[i], enc(i, r, 0), enc(i, r, 1), mask[i]);
      runConv(fmap[i], enc(i, r, 2), 3, Cl[i], Cl[i], Cpl[i], Cpl[i], Sl[i], Sl[i], Vl[i],
              mask[i], mask[i], nullptr, tmp);
      runBN(tmp, Vl[i], Cpl[i], Cl[i], enc(i, r, 3), enc(i, r, 4), mask[i]);
      runConv(tmp, enc(i, r, 5), 3, Cl[i], Cl[i], Cpl[i], Cpl[i], Sl[i], Sl[i], Vl[i], mask[i],
              mask[i], fmap[i], fmap[i]);
    }
    if (i < 4) {
      runBN(fmap[i], Vl[i], Cpl[i], Cl[i], dwn(i, 0), dwn(i, 1), mask[i]);
      k_maskpool<<<(Vl[i + 1] + 255) / 256, 256, 0, stream>>>(mask[i], mask[i + 1], Sl[i + 1]);
      runConv(fmap[i], dwn(i, 2), 2, Cl[i], Cl[i + 1], Cpl[i], Cpl[i + 1], Sl[i], Sl[i + 1],
              Vl[i + 1], mask[i], mask[i + 1], nullptr, fmap[i + 1]);
    }
  }

  k_gather_concat<<<(NPTS + 255) / 256, 256, 0, stream>>>(coords, fmap[0], fmap[1], fmap[2],
                                                          fmap[3], fmap[4], ptsA);

  const int Mtiles = NPTS / 16;  // 6250 exact
  auto runGemm = [&](int KC, const __bf16* Ab, float leak, __bf16* oB, float* oF,
                     const float* oBias, int Npad) {
    dim3 g((Mtiles + 7) / 8, (Npad + 31) / 32);
    switch (KC) {
      case 8:
        k_gemm_wmma<8><<<g, 256, 0, stream>>>(Ab, wrep, bnScale, bnBias, leak, oB, oF, oBias,
                                              Npad, Mtiles);
        break;
      case 6:
        k_gemm_wmma<6><<<g, 256, 0, stream>>>(Ab, wrep, bnScale, bnBias, leak, oB, oF, oBias,
                                              Npad, Mtiles);
        break;
      case 5:
        k_gemm_wmma<5><<<g, 256, 0, stream>>>(Ab, wrep, bnScale, bnBias, leak, oB, oF, oBias,
                                              Npad, Mtiles);
        break;
      case 3:
        k_gemm_wmma<3><<<g, 256, 0, stream>>>(Ab, wrep, bnScale, bnBias, leak, oB, oF, oBias,
                                              Npad, Mtiles);
        break;
      default:
        k_gemm_wmma<2><<<g, 256, 0, stream>>>(Ab, wrep, bnScale, bnBias, leak, oB, oF, oBias,
                                              Npad, Mtiles);
        break;
    }
  };

  const int Kp[4] = {256, 192, 160, 96};
  const int Np[4] = {192, 160, 96, 64};
  const int Kr[4] = {240, 192, 144, 96};
  const int Nr[4] = {192, 144, 96, 48};
  __bf16* cur = ptsA;
  __bf16* nxt = ptsB;
  for (int l = 0; l < 4; ++l) {
    runBN(cur, NPTS, Kp[l], Kr[l], dcw(l, 0), dcw(l, 1), nullptr);
    int total = Np[l] * Kp[l];
    k_repack<<<(total + 255) / 256, 256, 0, stream>>>(dcw(l, 2), wrep, 1, Kr[l], Nr[l], Kp[l],
                                                      Np[l]);
    runGemm(Kp[l] / 32, cur, 0.333f, nxt, nullptr, nullptr, Np[l]);
    __bf16* t2 = cur;
    cur = nxt;
    nxt = t2;
  }

  // final BN (relu) + 48->16 linear with bias, f32 output
  runBN(cur, NPTS, 64, 48, out_g, out_b, nullptr);
  k_repack<<<(16 * 64 + 255) / 256, 256, 0, stream>>>(lin_w, wrep, 1, 48, 16, 64, 16);
  runGemm(2, cur, 0.0f, nullptr, (float*)d_out, lin_b, 16);
}
